// ball_query_sample_with_goal_2662879723816
// MI455X (gfx1250) — compile-verified, matrix operand order checked by reading, not run on hardware
//
#include <hip/hip_runtime.h>
#include <hip/hip_bf16.h>
#include <math.h>
#include <stdint.h>

// ---------------------------------------------------------------------------
// CDNA5 (gfx1250) wave32 WMMA types/helpers
// ---------------------------------------------------------------------------
typedef __attribute__((ext_vector_type(16))) __bf16 v16bf;
typedef __attribute__((ext_vector_type(8)))  float  v8f;
typedef __attribute__((ext_vector_type(4)))  unsigned int v4ui;
typedef __attribute__((ext_vector_type(8)))  int v8i;
typedef __attribute__((ext_vector_type(4)))  int v4i;

union BF16x16 {
    v16bf v;
    unsigned short u[16];
    unsigned int   w[8];
};

__device__ __forceinline__ unsigned short f2bf(float f) {
    unsigned int x = __builtin_bit_cast(unsigned int, f);
    unsigned int r = x + 0x7FFFu + ((x >> 16) & 1u);   // round-to-nearest-even
    return (unsigned short)(r >> 16);
}

__device__ __forceinline__ v8f wmma_bf16(v16bf a, v16bf b, v8f c) {
    return __builtin_amdgcn_wmma_f32_16x16x32_bf16(
        false, a, false, b, (short)0, c, false, false);
}

// ---------------------------------------------------------------------------
// TDM: async 2D bf16 tile (tile_x elems wide, tile_rows tall) -> LDS,
// with 1-DWORD pad after every 16 DWORDs (=> 34-short LDS row pitch).
// D# bit layout per CDNA5 ISA ch.8 (group0: count/lds/global/type,
// group1: data_size=2B, pad_enable, dims, strides).
// ---------------------------------------------------------------------------
__device__ __forceinline__ void tdm_load_2d(
    unsigned lds_off, const void* gptr,
    unsigned tile_x, unsigned tile_rows,
    unsigned tensor_w, unsigned tensor_h, unsigned row_stride)
{
    unsigned long long ga = (unsigned long long)(uintptr_t)gptr;
    v4ui g0;
    g0[0] = 1u;                                              // count=1 (user D#)
    g0[1] = lds_off;                                         // lds_addr
    g0[2] = (unsigned)(ga & 0xffffffffu);                    // global_addr lo
    g0[3] = (unsigned)((ga >> 32) & 0x01ffffffu) | (2u << 30); // addr hi | type=2
    v8i g1;
    g1[0] = (int)((1u << 16) | (1u << 20) | (3u << 22));     // data_size=2B, pad_en, pad_int=16DW, pad_amt=1DW
    g1[1] = (int)((tensor_w & 0xffffu) << 16);               // tensor_dim0 lo16
    g1[2] = (int)(((tensor_w >> 16) & 0xffffu) | ((tensor_h & 0xffffu) << 16));
    g1[3] = (int)(((tensor_h >> 16) & 0xffffu) | ((tile_x & 0xffffu) << 16));
    g1[4] = (int)(tile_rows & 0xffffu);                      // tile_dim1 (tile_dim2=0)
    g1[5] = (int)row_stride;                                 // tensor_dim0_stride lo32
    g1[6] = 0;
    g1[7] = 0;
    v4i z4 = {0, 0, 0, 0};
#if __clang_major__ >= 23
    v8i z8 = {0, 0, 0, 0, 0, 0, 0, 0};
    __builtin_amdgcn_tensor_load_to_lds(g0, g1, z4, z4, z8, 0);
#else
    __builtin_amdgcn_tensor_load_to_lds(g0, g1, z4, z4, 0);
#endif
}

#define CC   256
#define NN   2048
#define BB   2
#define HH   8
#define HD   32
#define KQ   48
#define TOPK 32
#define RAD  32

// ---------------------------------------------------------------------------
// 0) transpose (B,C,N) -> x (B,N,C), f32 + bf16 copies
// ---------------------------------------------------------------------------
__global__ void transpose_kernel(const float* __restrict__ hf, float* __restrict__ x,
                                 unsigned short* __restrict__ xbf) {
    long id = (long)blockIdx.x * 256 + threadIdx.x;       // B*N*C
    if (id >= (long)BB * NN * CC) return;
    int c = id & 255;
    long nb = id >> 8;
    int n = (int)(nb % NN);
    int b = (int)(nb / NN);
    float v = hf[((size_t)b * CC + c) * NN + n];
    x[id] = v;
    xbf[id] = f2bf(v);
}

// one-shot weight convert: Wt[n][k] = bf16(W[k][n])   (pre-transposed for B-frags)
__global__ void convw_kernel(const float* __restrict__ W, unsigned short* __restrict__ Wt) {
    int id = blockIdx.x * 256 + threadIdx.x;   // CC*CC
    if (id >= CC * CC) return;
    int n = id >> 8, k = id & 255;
    Wt[id] = f2bf(W[(size_t)k * CC + n]);
}

// ---------------------------------------------------------------------------
// 1) bf16 WMMA GEMM (TDM-staged LDS tiles).
//    Y[M,256] = Xbf[M,256] @ Wt^T ;  outputs bf16 (Q pre-scaled by 1/sqrt(hd)).
// ---------------------------------------------------------------------------
__global__ __launch_bounds__(256) void gemm_qkv_kernel(
    const unsigned short* __restrict__ Xbf,
    const unsigned short* __restrict__ Wt0, const unsigned short* __restrict__ Wt1,
    const unsigned short* __restrict__ Wt2,
    unsigned short* __restrict__ O0, unsigned short* __restrict__ O1,
    unsigned short* __restrict__ O2, float qscale, int Mtot)
{
    const int Mbase = blockIdx.x * 64;
    const unsigned short* Wt = (blockIdx.y == 0) ? Wt0 : ((blockIdx.y == 1) ? Wt1 : Wt2);
    unsigned short* Out      = (blockIdx.y == 0) ? O0 : ((blockIdx.y == 1) ? O1 : O2);
    const float scl          = (blockIdx.y == 0) ? qscale : 1.0f;

    __shared__ unsigned short As[64][34];    // [m][k] (TDM-padded rows)
    __shared__ unsigned short Bs[256][34];   // [n][k]

    const int t = threadIdx.x;
    const int wave = t >> 5, lane = t & 31;
    const int mrow  = (wave & 3) * 16;
    const int nbase = (wave >> 2) * 128;
    const int lr = lane & 15;
    const int kb = (lane < 16) ? 0 : 8;

    v8f zero = {};
    v8f acc[8];
#pragma unroll
    for (int i = 0; i < 8; ++i) acc[i] = zero;

    for (int kt = 0; kt < 8; ++kt) {
        const int k0 = kt * 32;
        if (kt < 7)
            __builtin_prefetch(&Wt[(size_t)t * CC + k0 + 32], 0, 1);
        if (wave == 0) {
            tdm_load_2d((unsigned)(uintptr_t)&As[0][0],
                        Xbf + (size_t)Mbase * CC + k0, 32, 64, CC, (unsigned)Mtot, CC);
            tdm_load_2d((unsigned)(uintptr_t)&Bs[0][0],
                        Wt + k0, 32, 256, CC, CC, CC);
            __builtin_amdgcn_s_wait_tensorcnt(0);
        }
        __syncthreads();

        BF16x16 A;
        const int arow = mrow + lr;
#pragma unroll
        for (int p = 0; p < 8; ++p) {
            int kk = (p < 4) ? (kb + 2 * p) : (kb + 16 + 2 * (p - 4));
            A.w[p] = *(const unsigned int*)&As[arow][kk];
        }
#pragma unroll
        for (int nn2 = 0; nn2 < 8; ++nn2) {
            BF16x16 Bf;
            const int col = nbase + nn2 * 16 + lr;
#pragma unroll
            for (int p = 0; p < 8; ++p) {
                int kk = (p < 4) ? (kb + 2 * p) : (kb + 16 + 2 * (p - 4));
                Bf.w[p] = *(const unsigned int*)&Bs[col][kk];
            }
            acc[nn2] = wmma_bf16(A.v, Bf.v, acc[nn2]);
        }
        __syncthreads();
    }
#pragma unroll
    for (int nn2 = 0; nn2 < 8; ++nn2) {
        const int col = nbase + nn2 * 16 + lr;
#pragma unroll
        for (int j = 0; j < 8; ++j) {
            int row = Mbase + mrow + ((lane < 16) ? j : j + 8);
            Out[(size_t)row * CC + col] = f2bf(acc[nn2][j] * scl);
        }
    }
}

// GEMM with residual: x = res + o @ Wo ; writes f32 x and bf16 x copies
__global__ __launch_bounds__(256) void gemm_res_kernel(
    const unsigned short* __restrict__ Obf, const unsigned short* __restrict__ Wt,
    const float* __restrict__ Res, float* __restrict__ OutF,
    unsigned short* __restrict__ OutBf, int Mtot)
{
    const int Mbase = blockIdx.x * 64;
    __shared__ unsigned short As[64][34];
    __shared__ unsigned short Bs[256][34];
    const int t = threadIdx.x;
    const int wave = t >> 5, lane = t & 31;
    const int mrow  = (wave & 3) * 16;
    const int nbase = (wave >> 2) * 128;
    const int lr = lane & 15;
    const int kb = (lane < 16) ? 0 : 8;

    v8f zero = {};
    v8f acc[8];
#pragma unroll
    for (int i = 0; i < 8; ++i) acc[i] = zero;

    for (int kt = 0; kt < 8; ++kt) {
        const int k0 = kt * 32;
        if (kt < 7)
            __builtin_prefetch(&Wt[(size_t)t * CC + k0 + 32], 0, 1);
        if (wave == 0) {
            tdm_load_2d((unsigned)(uintptr_t)&As[0][0],
                        Obf + (size_t)Mbase * CC + k0, 32, 64, CC, (unsigned)Mtot, CC);
            tdm_load_2d((unsigned)(uintptr_t)&Bs[0][0],
                        Wt + k0, 32, 256, CC, CC, CC);
            __builtin_amdgcn_s_wait_tensorcnt(0);
        }
        __syncthreads();

        BF16x16 A;
        const int arow = mrow + lr;
#pragma unroll
        for (int p = 0; p < 8; ++p) {
            int kk = (p < 4) ? (kb + 2 * p) : (kb + 16 + 2 * (p - 4));
            A.w[p] = *(const unsigned int*)&As[arow][kk];
        }
#pragma unroll
        for (int nn2 = 0; nn2 < 8; ++nn2) {
            BF16x16 Bf;
            const int col = nbase + nn2 * 16 + lr;
#pragma unroll
            for (int p = 0; p < 8; ++p) {
                int kk = (p < 4) ? (kb + 2 * p) : (kb + 16 + 2 * (p - 4));
                Bf.w[p] = *(const unsigned int*)&Bs[col][kk];
            }
            acc[nn2] = wmma_bf16(A.v, Bf.v, acc[nn2]);
        }
        __syncthreads();
    }
#pragma unroll
    for (int nn2 = 0; nn2 < 8; ++nn2) {
        const int col = nbase + nn2 * 16 + lr;
#pragma unroll
        for (int j = 0; j < 8; ++j) {
            int row = Mbase + mrow + ((lane < 16) ? j : j + 8);
            float v = Res[(size_t)row * CC + col] + acc[nn2][j];
            OutF[(size_t)row * CC + col]  = v;
            OutBf[(size_t)row * CC + col] = f2bf(v);
        }
    }
}

// ---------------------------------------------------------------------------
// 2) Flash-style attention on bf16 Q/K/V (Q pre-scaled). One wave = 16 rows
//    of one (b,h). hd=32 == WMMA K. P transposed D->A via wave-private LDS.
// ---------------------------------------------------------------------------
__global__ __launch_bounds__(256) void attn_kernel(
    const unsigned short* __restrict__ Q, const unsigned short* __restrict__ Kv,
    const unsigned short* __restrict__ Vv, unsigned short* __restrict__ O)
{
    __shared__ unsigned short Ps[8][16][34];
    const int t = threadIdx.x;
    const int wave = t >> 5, lane = t & 31, lr = lane & 15;
    const int kb16 = (lane < 16) ? 0 : 8;

    const int tile = blockIdx.x * 8 + wave;   // 0..2047
    const int qt = tile & 127;
    const int h  = (tile >> 7) & 7;
    const int b  = tile >> 10;
    const size_t base = ((size_t)b * NN) * CC + (size_t)h * HD;

    BF16x16 Af;
    {
        const unsigned short* qp = Q + base + (size_t)(qt * 16 + lr) * CC;
#pragma unroll
        for (int p = 0; p < 8; ++p) {
            int d0 = (p < 4) ? (kb16 + 2 * p) : (kb16 + 16 + 2 * (p - 4));
            Af.w[p] = *(const unsigned int*)&qp[d0];
        }
    }

    v8f zero = {};
    v8f acc0 = zero, acc1 = zero;
    float mrun[8], srun[8];
#pragma unroll
    for (int j = 0; j < 8; ++j) { mrun[j] = -3.0e38f; srun[j] = 0.0f; }

    for (int kt = 0; kt < NN / 32; ++kt) {
        const int key0 = kt * 32;
        if (kt < NN / 32 - 1)
            __builtin_prefetch(&Kv[base + (size_t)(key0 + 32 + lr) * CC], 0, 1);

        BF16x16 K0, K1;
        {
            const unsigned short* kp0 = Kv + base + (size_t)(key0 + lr) * CC;
            const unsigned short* kp1 = Kv + base + (size_t)(key0 + 16 + lr) * CC;
#pragma unroll
            for (int p = 0; p < 8; ++p) {
                int d0 = (p < 4) ? (kb16 + 2 * p) : (kb16 + 16 + 2 * (p - 4));
                K0.w[p] = *(const unsigned int*)&kp0[d0];
                K1.w[p] = *(const unsigned int*)&kp1[d0];
            }
        }
        v8f s0 = wmma_bf16(Af.v, K0.v, zero);
        v8f s1 = wmma_bf16(Af.v, K1.v, zero);

#pragma unroll
        for (int j = 0; j < 8; ++j) {
            float mx = fmaxf(s0[j], s1[j]);
#pragma unroll
            for (int d_ = 1; d_ < 16; d_ <<= 1) mx = fmaxf(mx, __shfl_xor(mx, d_, 16));
            float mnew = fmaxf(mrun[j], mx);
            float corr = __expf(mrun[j] - mnew);
            float p0 = __expf(s0[j] - mnew);
            float p1 = __expf(s1[j] - mnew);
            float rs = p0 + p1;
#pragma unroll
            for (int d_ = 1; d_ < 16; d_ <<= 1) rs += __shfl_xor(rs, d_, 16);
            srun[j] = srun[j] * corr + rs;
            mrun[j] = mnew;
            acc0[j] *= corr;
            acc1[j] *= corr;
            const int prow = (lane < 16) ? j : j + 8;
            Ps[wave][prow][lr]      = f2bf(p0);
            Ps[wave][prow][16 + lr] = f2bf(p1);
        }

        BF16x16 Pf;
#pragma unroll
        for (int p = 0; p < 8; ++p) {
            int kk = (p < 4) ? (kb16 + 2 * p) : (kb16 + 16 + 2 * (p - 4));
            Pf.w[p] = *(const unsigned int*)&Ps[wave][lr][kk];
        }
        BF16x16 V0, V1;
#pragma unroll
        for (int p = 0; p < 8; ++p) {
            int kk = (p < 4) ? (kb16 + 2 * p) : (kb16 + 16 + 2 * (p - 4));
            const unsigned short* vp0 = Vv + base + (size_t)(key0 + kk) * CC;
            const unsigned short* vp1 = Vv + base + (size_t)(key0 + kk + 1) * CC;
            V0.w[p] = (unsigned)vp0[lr]      | ((unsigned)vp1[lr] << 16);
            V1.w[p] = (unsigned)vp0[16 + lr] | ((unsigned)vp1[16 + lr] << 16);
        }
        acc0 = wmma_bf16(Pf.v, V0.v, acc0);
        acc1 = wmma_bf16(Pf.v, V1.v, acc1);
    }

#pragma unroll
    for (int j = 0; j < 8; ++j) {
        const int m = qt * 16 + ((lane < 16) ? j : j + 8);
        const float inv = 1.0f / srun[j];
        unsigned short* op = O + base + (size_t)m * CC;
        op[lr]      = f2bf(acc0[j] * inv);
        op[16 + lr] = f2bf(acc1[j] * inv);
    }
}

// ---------------------------------------------------------------------------
// 3) generic linear + BN + ReLU (f-net)
// ---------------------------------------------------------------------------
__global__ void lin_kernel(const float* __restrict__ X, const float* __restrict__ W,
                           const float* __restrict__ bias, float* __restrict__ Y,
                           int Cin, int Cout, int N, long sb, long sc, long sn, long total)
{
    long id = (long)blockIdx.x * 256 + threadIdx.x;
    if (id >= total) return;
    int n = (int)(id % N);
    int o = (int)((id / N) % Cout);
    int b = (int)(id / ((long)N * Cout));
    const float* xp = X + (long)b * sb + (long)n * sn;
    const float* wp = W + (size_t)o * Cin;
    float acc = bias[o];
    for (int c = 0; c < Cin; ++c) acc += wp[c] * xp[(long)c * sc];
    Y[id] = acc;
}

__global__ __launch_bounds__(256) void bnstats_kernel(
    const float* __restrict__ Y, float* __restrict__ mean, float* __restrict__ var,
    int B_, int Cc, int R)
{
    const int ch = blockIdx.x;
    const int t = threadIdx.x;
    float s = 0.f, s2 = 0.f;
    for (int b = 0; b < B_; ++b) {
        const float* p = Y + ((size_t)b * Cc + ch) * R;
        for (int r = t; r < R; r += 256) { float v = p[r]; s += v; s2 += v * v; }
    }
    __shared__ float sh[256], sh2[256];
    sh[t] = s; sh2[t] = s2; __syncthreads();
    for (int st = 128; st > 0; st >>= 1) {
        if (t < st) { sh[t] += sh[t + st]; sh2[t] += sh2[t + st]; }
        __syncthreads();
    }
    if (t == 0) {
        float cnt = (float)B_ * (float)R;
        float m = sh[0] / cnt;
        mean[ch] = m;
        var[ch]  = sh2[0] / cnt - m * m;
    }
}

__global__ void bnrelu_kernel(float* __restrict__ Y, const float* __restrict__ mean,
                              const float* __restrict__ var, const float* __restrict__ g,
                              const float* __restrict__ be, int Cc, int R, long total)
{
    long id = (long)blockIdx.x * 256 + threadIdx.x;
    if (id >= total) return;
    int ch = (int)((id / R) % Cc);
    float v = Y[id];
    float o = g[ch] * (v - mean[ch]) * rsqrtf(var[ch] + 1e-5f) + be[ch];
    Y[id] = (o > 0.f) ? o : 0.f;
}

// ---------------------------------------------------------------------------
// 4) top-k (k=32, sorted desc, tie -> lowest index) per (b, channel)
// ---------------------------------------------------------------------------
__global__ __launch_bounds__(256) void topk_kernel(const float* __restrict__ H,
                                                   int* __restrict__ idxout, int N)
{
    const int row = blockIdx.x;
    const int t = threadIdx.x;
    __shared__ float vals[NN];
    __shared__ float rv[256];
    __shared__ int   ri[256];
    const float* src = H + (size_t)row * N;
    for (int i = t; i < N; i += 256) vals[i] = src[i];
    __syncthreads();
    for (int j = 0; j < TOPK; ++j) {
        float bv = -3.4e38f; int bi = 0x7fffffff;
        for (int i = t; i < N; i += 256) {
            float v = vals[i];
            if (v > bv || (v == bv && i < bi)) { bv = v; bi = i; }
        }
        rv[t] = bv; ri[t] = bi; __syncthreads();
        for (int s = 128; s > 0; s >>= 1) {
            if (t < s) {
                float v2 = rv[t + s]; int i2 = ri[t + s];
                if (v2 > rv[t] || (v2 == rv[t] && i2 < ri[t])) { rv[t] = v2; ri[t] = i2; }
            }
            __syncthreads();
        }
        if (t == 0) { idxout[(size_t)row * TOPK + j] = ri[0]; vals[ri[0]] = -3.4e38f; }
        __syncthreads();
    }
}

__global__ void gather_query_kernel(const float* __restrict__ pts,
                                    const int* __restrict__ idx32,
                                    float* __restrict__ qpts, float* __restrict__ outres, int N)
{
    int i = blockIdx.x * blockDim.x + threadIdx.x;
    if (i >= BB * KQ) return;
    int b = i / KQ;
    int n = idx32[(size_t)i * TOPK];
    for (int d = 0; d < 3; ++d) {
        float v = pts[(size_t)b * 3 * N + (size_t)d * N + n];
        qpts[i * 3 + d] = v;
        outres[i * 3 + d] = v;
    }
}

__global__ __launch_bounds__(256) void ball_kernel(const float* __restrict__ pts,
                                                   const float* __restrict__ qpts,
                                                   float* __restrict__ rp, int N)
{
    const int row = blockIdx.x;
    const int b = row / KQ;
    const int t = threadIdx.x;
    __shared__ float vals[NN];
    __shared__ float rv[256];
    __shared__ int   ri[256];
    __shared__ int   sel[RAD];
    const float qx = qpts[row * 3 + 0], qy = qpts[row * 3 + 1], qz = qpts[row * 3 + 2];
    const float* px = pts + (size_t)b * 3 * N;
    for (int i = t; i < N; i += 256) {
        float dx = px[i] - qx, dy = px[N + i] - qy, dz = px[2 * N + i] - qz;
        vals[i] = -(dx * dx + dy * dy + dz * dz);
    }
    __syncthreads();
    for (int j = 0; j < RAD; ++j) {
        float bv = -3.4e38f; int bi = 0x7fffffff;
        for (int i = t; i < N; i += 256) {
            float v = vals[i];
            if (v > bv || (v == bv && i < bi)) { bv = v; bi = i; }
        }
        rv[t] = bv; ri[t] = bi; __syncthreads();
        for (int s = 128; s > 0; s >>= 1) {
            if (t < s) {
                float v2 = rv[t + s]; int i2 = ri[t + s];
                if (v2 > rv[t] || (v2 == rv[t] && i2 < ri[t])) { rv[t] = v2; ri[t] = i2; }
            }
            __syncthreads();
        }
        if (t == 0) { sel[j] = ri[0]; vals[ri[0]] = -3.4e38f; }
        __syncthreads();
    }
    if (t < RAD) {
        int n = sel[t];
        rp[((size_t)row * RAD + t) * 3 + 0] = px[n];
        rp[((size_t)row * RAD + t) * 3 + 1] = px[N + n];
        rp[((size_t)row * RAD + t) * 3 + 2] = px[2 * N + n];
    }
}

// ---------------------------------------------------------------------------
// 7) r-net head
// ---------------------------------------------------------------------------
__global__ void rnet1_kernel(const float* __restrict__ rp, const float* __restrict__ w1,
                             const float* __restrict__ b1, float* __restrict__ y1)
{
    int id = blockIdx.x * 256 + threadIdx.x;
    if (id >= BB * 128 * KQ * RAD) return;
    int m  = id & 31;
    int kq = (id >> 5) % KQ;
    int o  = (id / (RAD * KQ)) % 128;
    int b  = id / (RAD * KQ * 128);
    const float* p = rp + (((size_t)(b * KQ + kq) * RAD) + m) * 3;
    y1[id] = b1[o] + w1[o * 3] * p[0] + w1[o * 3 + 1] * p[1] + w1[o * 3 + 2] * p[2];
}

__global__ void rnet2_kernel(const float* __restrict__ y1v, const float* __restrict__ w2,
                             const float* __restrict__ b2, float* __restrict__ y2v)
{
    long id = (long)blockIdx.x * 256 + threadIdx.x;
    if (id >= (long)BB * 256 * KQ * RAD) return;
    int r = (int)(id % (KQ * RAD));
    int o = (int)((id / (KQ * RAD)) % 256);
    int b = (int)(id / ((long)KQ * RAD * 256));
    const float* xp = y1v + (size_t)b * 128 * (KQ * RAD) + r;
    const float* wp = w2 + (size_t)o * 128;
    float acc = b2[o];
    for (int c = 0; c < 128; ++c) acc += wp[c] * xp[(size_t)c * (KQ * RAD)];
    y2v[id] = acc;
}

__global__ void rnet3_kernel(const float* __restrict__ y2v, const float* __restrict__ w3,
                             const float* __restrict__ b3, float* __restrict__ y3v)
{
    int id = blockIdx.x * 256 + threadIdx.x;
    if (id >= BB * 480 * KQ) return;
    int kq = id % KQ;
    int o  = (id / KQ) % 480;
    int b  = id / (KQ * 480);
    const float* xp = y2v + (size_t)b * 256 * (KQ * RAD) + kq * RAD;
    const float* wp = w3 + (size_t)o * 256 * RAD;
    float acc = b3[o];
    for (int c = 0; c < 256; ++c)
        for (int m = 0; m < RAD; ++m)
            acc += wp[c * RAD + m] * xp[(size_t)c * (KQ * RAD) + m];
    y3v[id] = acc;
}

__global__ void final_kernel(const float* __restrict__ y3v, const int* __restrict__ idx32,
                             float* __restrict__ outp)
{
    int id = blockIdx.x * 256 + threadIdx.x;
    if (id >= BB * 512 * KQ) return;
    int k  = id % KQ;
    int ch = (id / KQ) % 512;
    int b  = id / (KQ * 512);
    float v;
    if (ch < 480) v = y3v[((size_t)(b * 480 + ch)) * KQ + k];
    else          v = (float)idx32[((size_t)(b * KQ + k)) * TOPK + (ch - 480)];
    outp[id] = v;
}

// ---------------------------------------------------------------------------
// host orchestration
// ---------------------------------------------------------------------------
extern "C" void kernel_launch(void* const* d_in, const int* in_sizes, int n_in,
                              void* d_out, int out_size, void* d_ws, size_t ws_size,
                              hipStream_t stream)
{
    const float* hf    = (const float*)d_in[0];
    const float* inp   = (const float*)d_in[1];
    const float* sa_w[4] = {(const float*)d_in[4], (const float*)d_in[5],
                            (const float*)d_in[6], (const float*)d_in[7]};
    const float* f_w[3]  = {(const float*)d_in[8],  (const float*)d_in[9],  (const float*)d_in[10]};
    const float* f_b[3]  = {(const float*)d_in[11], (const float*)d_in[12], (const float*)d_in[13]};
    const float* f_g[3]  = {(const float*)d_in[14], (const float*)d_in[15], (const float*)d_in[16]};
    const float* f_be[3] = {(const float*)d_in[17], (const float*)d_in[18], (const float*)d_in[19]};
    const float* r_w1 = (const float*)d_in[20];
    const float* r_b1 = (const float*)d_in[21];
    const float* r_g1 = (const float*)d_in[22];
    const float* r_be1= (const float*)d_in[23];
    const float* r_w2 = (const float*)d_in[24];
    const float* r_b2 = (const float*)d_in[25];
    const float* r_g2 = (const float*)d_in[26];
    const float* r_be2= (const float*)d_in[27];
    const float* r_w3 = (const float*)d_in[28];
    const float* r_b3 = (const float*)d_in[29];
    const float* r_g3 = (const float*)d_in[30];
    const float* r_be3= (const float*)d_in[31];

    // ---- workspace layout (byte cursor) ----
    char* ws = (char*)d_ws;
    float* xbuf = (float*)ws;              ws += (size_t)BB * NN * CC * 4;   // f32 x
    unsigned short* xbf = (unsigned short*)ws; ws += (size_t)BB * NN * CC * 2;
    unsigned short* qbf = (unsigned short*)ws; ws += (size_t)BB * NN * CC * 2;
    unsigned short* kbf = (unsigned short*)ws; ws += (size_t)BB * NN * CC * 2;
    unsigned short* vbf = (unsigned short*)ws; ws += (size_t)BB * NN * CC * 2;
    unsigned short* obf = (unsigned short*)ws; ws += (size_t)BB * NN * CC * 2;
    unsigned short* wt  = (unsigned short*)ws; ws += (size_t)8 * CC * CC * 2; // 8 bf16 W^T
    float* h1   = (float*)ws;              ws += (size_t)BB * 64 * NN * 4;
    float* h2   = (float*)ws;              ws += (size_t)BB * 64 * NN * 4;
    float* h3   = (float*)ws;              ws += (size_t)BB * KQ * NN * 4;
    float* meanb= (float*)ws;              ws += 512 * 4;
    float* varb = (float*)ws;              ws += 512 * 4;
    int*   idx32= (int*)ws;                ws += (size_t)BB * KQ * TOPK * 4;
    float* qpts = (float*)ws;              ws += (size_t)BB * KQ * 3 * 4;
    float* rp   = (float*)ws;              ws += (size_t)BB * KQ * RAD * 3 * 4;
    float* y1   = (float*)ws;              ws += (size_t)BB * 128 * KQ * RAD * 4;
    float* y2   = (float*)ws;              ws += (size_t)BB * 256 * KQ * RAD * 4;
    float* y3   = (float*)ws;              ws += (size_t)BB * 480 * KQ * 4;

    float* out_main = (float*)d_out;
    float* out_res  = out_main + BB * 512 * KQ;

    const float qscale = 0.17677669529663687f;   // 1/sqrt(32)
    const int   Mtot = BB * NN;                  // 4096
    const long  MNC = (long)BB * NN * CC;

    // x = hf^T (f32 + bf16); weights -> bf16 transposed (one-shot)
    transpose_kernel<<<(MNC + 255) / 256, 256, 0, stream>>>(hf, xbuf, xbf);
    for (int l = 0; l < 2; ++l)
        for (int wmi = 0; wmi < 4; ++wmi)
            convw_kernel<<<(CC * CC + 255) / 256, 256, 0, stream>>>(
                sa_w[wmi] + (size_t)l * CC * CC, wt + (size_t)(l * 4 + wmi) * CC * CC);

    // 2 self-attention layers (TDM-staged bf16 WMMA)
    for (int l = 0; l < 2; ++l) {
        unsigned short* wq = wt + (size_t)(l * 4 + 0) * CC * CC;
        unsigned short* wk = wt + (size_t)(l * 4 + 1) * CC * CC;
        unsigned short* wv = wt + (size_t)(l * 4 + 2) * CC * CC;
        unsigned short* wo = wt + (size_t)(l * 4 + 3) * CC * CC;
        gemm_qkv_kernel<<<dim3(64, 3), 256, 0, stream>>>(
            xbf, wq, wk, wv, qbf, kbf, vbf, qscale, Mtot);
        attn_kernel<<<256, 256, 0, stream>>>(qbf, kbf, vbf, obf);
        gemm_res_kernel<<<64, 256, 0, stream>>>(obf, wo, xbuf, xbuf, xbf, Mtot);
    }

    // f-net: 256 -> 64 -> 64 -> 48 with BN + ReLU
    {
        long tot = (long)BB * 64 * NN;
        lin_kernel<<<(tot + 255) / 256, 256, 0, stream>>>(xbuf, f_w[0], f_b[0], h1,
            256, 64, NN, (long)NN * CC, 1L, (long)CC, tot);
        bnstats_kernel<<<64, 256, 0, stream>>>(h1, meanb, varb, BB, 64, NN);
        bnrelu_kernel<<<(tot + 255) / 256, 256, 0, stream>>>(h1, meanb, varb, f_g[0], f_be[0], 64, NN, tot);

        lin_kernel<<<(tot + 255) / 256, 256, 0, stream>>>(h1, f_w[1], f_b[1], h2,
            64, 64, NN, (long)64 * NN, (long)NN, 1L, tot);
        bnstats_kernel<<<64, 256, 0, stream>>>(h2, meanb, varb, BB, 64, NN);
        bnrelu_kernel<<<(tot + 255) / 256, 256, 0, stream>>>(h2, meanb, varb, f_g[1], f_be[1], 64, NN, tot);

        long tot3 = (long)BB * KQ * NN;
        lin_kernel<<<(tot3 + 255) / 256, 256, 0, stream>>>(h2, f_w[2], f_b[2], h3,
            64, KQ, NN, (long)64 * NN, (long)NN, 1L, tot3);
        bnstats_kernel<<<KQ, 256, 0, stream>>>(h3, meanb, varb, BB, KQ, NN);
        bnrelu_kernel<<<(tot3 + 255) / 256, 256, 0, stream>>>(h3, meanb, varb, f_g[2], f_be[2], KQ, NN, tot3);
    }

    // selection + grouping
    topk_kernel<<<BB * KQ, 256, 0, stream>>>(h3, idx32, NN);
    gather_query_kernel<<<1, 128, 0, stream>>>(inp, idx32, qpts, out_res, NN);
    ball_kernel<<<BB * KQ, 256, 0, stream>>>(inp, qpts, rp, NN);

    // r-net head
    {
        long t1 = (long)BB * 128 * KQ * RAD;
        rnet1_kernel<<<(t1 + 255) / 256, 256, 0, stream>>>(rp, r_w1, r_b1, y1);
        bnstats_kernel<<<128, 256, 0, stream>>>(y1, meanb, varb, BB, 128, KQ * RAD);
        bnrelu_kernel<<<(t1 + 255) / 256, 256, 0, stream>>>(y1, meanb, varb, r_g1, r_be1, 128, KQ * RAD, t1);

        long t2 = (long)BB * 256 * KQ * RAD;
        rnet2_kernel<<<(t2 + 255) / 256, 256, 0, stream>>>(y1, r_w2, r_b2, y2);
        bnstats_kernel<<<256, 256, 0, stream>>>(y2, meanb, varb, BB, 256, KQ * RAD);
        bnrelu_kernel<<<(t2 + 255) / 256, 256, 0, stream>>>(y2, meanb, varb, r_g2, r_be2, 256, KQ * RAD, t2);

        long t3 = (long)BB * 480 * KQ;
        rnet3_kernel<<<(t3 + 255) / 256, 256, 0, stream>>>(y2, r_w3, r_b3, y3);
        bnstats_kernel<<<480, 256, 0, stream>>>(y3, meanb, varb, BB, 480, KQ);
        bnrelu_kernel<<<(t3 + 255) / 256, 256, 0, stream>>>(y3, meanb, varb, r_g3, r_be3, 480, KQ, t3);
    }

    long tf = (long)BB * 512 * KQ;
    final_kernel<<<(tf + 255) / 256, 256, 0, stream>>>(y3, idx32, out_main);
}
